// OAdder2d_Q_33818572489494
// MI455X (gfx1250) — matrix operand order repaired and checked by reading, not compiled, not run on hardware
//
#include <hip/hip_runtime.h>
#include <hip/hip_bf16.h>
#include <math.h>

typedef _Float16 v8h  __attribute__((ext_vector_type(8)));
typedef _Float16 v16h __attribute__((ext_vector_type(16)));
typedef float    v8f  __attribute__((ext_vector_type(8)));
typedef float    v4f  __attribute__((ext_vector_type(4)));

#define BATCH 32
#define CIN   128
#define COUT  256
#define HH    56
#define WWID  56
#define HWSZ  (HH*WWID)            // 3136
#define KTOT  (CIN*9)              // 1152
#define MTOT  (BATCH*HWSZ)         // 100352
#define NW    (COUT*KTOT)          // 294912 weight elems
#define NX    (BATCH*CIN*HWSZ)     // 12845056 x elems
#define QLV   65535.0f
#define LDSTR 40                   // LDS row stride in halves (80B, 16B-aligned rows)

// workspace layout (bytes)
#define XQ_OFF    0u               // xq: NHWC f16 [B][HW][C]
#define KEFF_OFF  25690112u        // NX*2
#define SQ_OFF    26279936u        // +NW*2
#define WT_OFF    26681344u        // +MTOT*4
#define COEF_OFF  26682368u        // +COUT*4  (coef stored rs-major [rs*128+c])
#define GMAX_OFF  26686976u        // +KTOT*4

__global__ void k_init(float* gmax) {
    if (blockIdx.x == 0 && threadIdx.x == 0) *gmax = 0.0f;
}

__global__ void k_absmax(const float* __restrict__ w, unsigned int* gmax, int n) {
    __shared__ float red[256];
    int i = blockIdx.x * 256 + threadIdx.x;
    float v = 0.0f;
    if (i < n) v = fabsf(tanhf(w[i]));
    red[threadIdx.x] = v;
    __syncthreads();
    for (int s = 128; s > 0; s >>= 1) {
        if (threadIdx.x < s) {
            float o = red[threadIdx.x + s];
            if (o > red[threadIdx.x]) red[threadIdx.x] = o;
        }
        __syncthreads();
    }
    if (threadIdx.x == 0) atomicMax(gmax, __float_as_uint(red[0]));
}

// coef permuted rs-major: coef[rs*128 + c] = (d0-d1)/4 at natural (c,r,s)
__global__ void k_coef(const float* __restrict__ disks, float* __restrict__ coef) {
    int i = blockIdx.x * blockDim.x + threadIdx.x;
    if (i >= KTOT) return;
    int rs  = i >> 7;
    int c   = i & 127;
    int crs = c * 9 + rs;
    coef[i] = (disks[2*crs] - disks[2*crs+1]) * 0.25f;
}

// keff permuted to [O][rs*128 + c]; i indexes the DESTINATION.
__global__ void k_prepw(const float* __restrict__ w, const float* __restrict__ phases,
                        const float* __restrict__ disks, const float* __restrict__ gmax,
                        _Float16* __restrict__ keff, int n) {
    int i = blockIdx.x * blockDim.x + threadIdx.x;
    if (i >= n) return;
    int o   = i / KTOT;
    int rem = i - o * KTOT;
    int rs  = rem >> 7;
    int c   = rem & 127;
    int crs = c * 9 + rs;                       // natural (C,3,3) index
    float g  = *gmax;
    float t  = tanhf(w[o * KTOT + crs]);
    float wq = rintf((t / (2.0f * g) + 0.5f) * QLV) / QLV;
    float s  = sinf(phases[crs]);
    float dm = 0.5f * (disks[2*crs] + disks[2*crs+1]);
    keff[i] = (_Float16)(wq * s * dm);
}

__global__ void k_wterm(const float* __restrict__ w, const float* __restrict__ gmax,
                        const float* __restrict__ coef, float* __restrict__ wt) {
    int o = threadIdx.x;  // COUT == 256 == blockDim
    float g = *gmax;
    float acc = 0.0f;
    for (int j = 0; j < KTOT; ++j) {            // j in rs-major order
        int rs = j >> 7;
        int c  = j & 127;
        float t  = tanhf(w[o * KTOT + c * 9 + rs]);
        float wq = rintf((t / (2.0f * g) + 0.5f) * QLV) / QLV;
        acc += wq * wq * coef[j];
    }
    wt[o] = acc;
}

// quantize + NCHW -> NHWC transpose through LDS.
// grid: x = 49 p-tiles * 4 c-tiles = 196, y = batch(32); 256 threads.
__global__ void k_quantx(const float* __restrict__ x, _Float16* __restrict__ xq) {
    __shared__ _Float16 tile[64][33];
    const int bidx  = blockIdx.y;
    const int ptile = blockIdx.x >> 2;
    const int ctile = blockIdx.x & 3;
    const int p0 = ptile * 64;
    const int c0 = ctile * 32;
    const int t  = threadIdx.x;
    {   // read 32 c x 64 p, coalesced along p
        int cl = t >> 3;              // 0..31
        int pb = (t & 7) * 8;         // 0..56
        const float* src = x + ((size_t)(bidx * CIN + c0 + cl) * HWSZ) + p0 + pb;
        #pragma unroll
        for (int j = 0; j < 8; ++j) {
            float v = src[j];
            v = v < 0.0f ? 0.0f : (v > 1.0f ? 1.0f : v);
            tile[pb + j][cl] = (_Float16)(rintf(v * QLV) / QLV);
        }
    }
    __syncthreads();
    {   // write 64 p x 32 c, 16B chunks along c
        int pl = t >> 2;              // 0..63
        int cb = (t & 3) * 8;         // 0,8,16,24
        _Float16* dst = xq + ((size_t)bidx * HWSZ + p0 + pl) * CIN + c0 + cb;
        v8h v;
        #pragma unroll
        for (int j = 0; j < 8; ++j) v[j] = tile[pl][cb + j];
        *(v8h*)dst = v;
    }
}

// sq[m] = sum_{c,rs} xq^2 * coef  (NHWC x, rs-major coef -> contiguous inner loop)
__global__ void k_sq(const _Float16* __restrict__ xq, const float* __restrict__ coef,
                     float* __restrict__ sq) {
    int m = blockIdx.x * 256 + threadIdx.x;
    if (m >= MTOT) return;
    int b  = m / HWSZ;
    int hw = m - b * HWSZ;
    int h  = hw / WWID;
    int wc = hw - h * WWID;
    const _Float16* xb = xq + (size_t)b * HWSZ * CIN;
    float acc = 0.0f;
    #pragma unroll
    for (int r = 0; r < 3; ++r) {
        int h2 = h + r - 1;
        if ((unsigned)h2 >= HH) continue;
        #pragma unroll
        for (int s = 0; s < 3; ++s) {
            int w2 = wc + s - 1;
            if ((unsigned)w2 >= WWID) continue;
            const _Float16* xp = xb + (size_t)(h2 * WWID + w2) * CIN;
            const float*    cp = coef + (r * 3 + s) * CIN;
            #pragma unroll 4
            for (int c = 0; c < CIN; ++c) {
                float xv = (float)xp[c];
                acc += xv * xv * cp[c];
            }
        }
    }
    sq[m] = acc;
}

__launch_bounds__(256)
__global__ void k_gemm(const _Float16* __restrict__ xq, const _Float16* __restrict__ keff,
                       const float* __restrict__ sq, const float* __restrict__ wt,
                       float* __restrict__ out) {
    __shared__ __align__(16) _Float16 Al[128 * LDSTR];
    __shared__ __align__(16) _Float16 Bl[128 * LDSTR];

    const int t    = threadIdx.x;
    const int lane = t & 31;
    const int wv   = t >> 5;
    const int wm   = wv & 3;     // wave M position (0..3) -> 32 rows each
    const int wn   = wv >> 2;    // wave N position (0..1) -> 64 cols each
    const int mblk = blockIdx.x * 128;
    const int nblk = blockIdx.y * 128;

    // lane-derived fragment addressing (contiguous k = 16*hi + s, same order for A and B)
    const int lm   = lane & 15;
    const int hi   = lane >> 4;
    const int kloA = hi * 16;      // A: halves K=kloA..+15
    const int kloB = hi * 16;      // B: halves K=kloB..+15

    // A-gather mapping: 128 m-columns, each thread owns 16 CONTIGUOUS kk
    const int tm  = t & 127;
    const int tk0 = (t >> 7) * 16;             // kk base: 0 or 16
    const int m   = mblk + tm;                 // MTOT divisible by 128 -> always valid
    const int bi  = m / HWSZ;
    const int hw  = m - bi * HWSZ;
    const int hh  = hw / WWID;
    const int wc  = hw - hh * WWID;
    const _Float16* xbase = xq + (size_t)bi * HWSZ * CIN;   // NHWC batch base

    v8f acc[2][4];
    #pragma unroll
    for (int i = 0; i < 2; ++i)
        #pragma unroll
        for (int j = 0; j < 4; ++j)
            acc[i][j] = {};

    for (int kt = 0; kt < KTOT / 32; ++kt) {
        // k = rs*128 + c  ->  this k-step has scalar (r,s), channels c0..c0+31
        const int rsIdx = kt >> 2;
        const int c0    = (kt & 3) * 32;
        const int r     = rsIdx / 3;
        const int s     = rsIdx - r * 3;
        const int h2    = hh + r - 1;
        const int w2    = wc + s - 1;
        const bool inb  = ((unsigned)h2 < HH) & ((unsigned)w2 < WWID);

        // ---- stage A tile: NHWC -> channels contiguous: two b128 loads ----
        v8h alo = {}, ahi = {};
        if (inb) {
            const v8h* src = (const v8h*)(xbase + (size_t)(h2 * WWID + w2) * CIN + c0 + tk0);
            alo = src[0];
            ahi = src[1];
        }
        *(v8h*)(&Al[tm * LDSTR + tk0])     = alo;
        *(v8h*)(&Al[tm * LDSTR + tk0 + 8]) = ahi;

        // ---- stage B tile (128 N-rows x 32 K, contiguous copies) ----
        const int kb = kt * 32;
        #pragma unroll
        for (int p = 0; p < 2; ++p) {
            int chunk = t + p * 256;
            int n     = chunk >> 2;
            int off   = (chunk & 3) * 8;
            const v8h* src = (const v8h*)(keff + (size_t)(nblk + n) * KTOT + kb + off);
            *(v8h*)(&Bl[n * LDSTR + off]) = *src;
        }
        __syncthreads();

        // ---- fragments ----
        v16h afr[2];
        #pragma unroll
        for (int mi = 0; mi < 2; ++mi) {
            const v8h* p8 = (const v8h*)&Al[(wm * 32 + mi * 16 + lm) * LDSTR + kloA];
            afr[mi] = __builtin_shufflevector(p8[0], p8[1],
                        0,1,2,3,4,5,6,7,8,9,10,11,12,13,14,15);
        }
        v16h bfr[4];
        #pragma unroll
        for (int ni = 0; ni < 4; ++ni) {
            const v8h* q8 = (const v8h*)&Bl[(wn * 64 + ni * 16 + lm) * LDSTR + kloB];
            bfr[ni] = __builtin_shufflevector(q8[0], q8[1],
                        0,1,2,3,4,5,6,7,8,9,10,11,12,13,14,15);
        }
        // ---- 8 WMMAs per wave per k-step ----
        #pragma unroll
        for (int mi = 0; mi < 2; ++mi)
            #pragma unroll
            for (int ni = 0; ni < 4; ++ni)
                acc[mi][ni] = __builtin_amdgcn_wmma_f32_16x16x32_f16(
                    false, afr[mi], false, bfr[ni],
                    (short)0, acc[mi][ni], false, false);
        __syncthreads();
    }

    // ---- epilogue: out[m][n] = gemm + sq[m] + wt[n] ----
    #pragma unroll
    for (int mi = 0; mi < 2; ++mi) {
        int mbase = mblk + wm * 32 + mi * 16 + hi * 8;
        int b2    = mbase / HWSZ;
        int rem   = mbase - b2 * HWSZ;
        float sqv[8];
        #pragma unroll
        for (int j = 0; j < 8; ++j) sqv[j] = sq[mbase + j];
        #pragma unroll
        for (int ni = 0; ni < 4; ++ni) {
            int n = nblk + wn * 64 + ni * 16 + lm;
            float wtv = wt[n];
            #pragma unroll
            for (int j = 0; j < 8; ++j) {
                int hw2 = rem + j;
                int bb  = b2;
                if (hw2 >= HWSZ) { bb += 1; hw2 -= HWSZ; }
                out[((size_t)bb * COUT + n) * HWSZ + hw2] = acc[mi][ni][j] + sqv[j] + wtv;
            }
        }
    }
}

extern "C" void kernel_launch(void* const* d_in, const int* in_sizes, int n_in,
                              void* d_out, int out_size, void* d_ws, size_t ws_size,
                              hipStream_t stream) {
    (void)in_sizes; (void)n_in; (void)out_size; (void)ws_size;
    const float* x      = (const float*)d_in[0];
    const float* weight = (const float*)d_in[1];
    const float* phases = (const float*)d_in[2];
    const float* disks  = (const float*)d_in[3];

    char* ws = (char*)d_ws;
    _Float16* xq   = (_Float16*)(ws + XQ_OFF);
    _Float16* keff = (_Float16*)(ws + KEFF_OFF);
    float*    sqb  = (float*)(ws + SQ_OFF);
    float*    wtb  = (float*)(ws + WT_OFF);
    float*    coef = (float*)(ws + COEF_OFF);
    float*    gmax = (float*)(ws + GMAX_OFF);
    float*    out  = (float*)d_out;

    k_init  <<<1, 1, 0, stream>>>(gmax);
    k_absmax<<<(NW + 255) / 256, 256, 0, stream>>>(weight, (unsigned int*)gmax, NW);
    k_coef  <<<(KTOT + 255) / 256, 256, 0, stream>>>(disks, coef);
    k_prepw <<<(NW + 255) / 256, 256, 0, stream>>>(weight, phases, disks, gmax, keff, NW);
    k_wterm <<<1, 256, 0, stream>>>(weight, gmax, coef, wtb);
    k_quantx<<<dim3(196, BATCH), 256, 0, stream>>>(x, xq);
    k_sq    <<<(MTOT + 255) / 256, 256, 0, stream>>>(xq, coef, sqb);
    k_gemm  <<<dim3(MTOT / 128, COUT / 128), 256, 0, stream>>>(xq, keff, sqb, wtb, out);
}
